// MultiHeadAttention_15839839387691
// MI455X (gfx1250) — matrix operand order repaired and checked by reading, not compiled, not run on hardware
//
#include <hip/hip_runtime.h>
#include <hip/hip_bf16.h>

#define D_MODEL 768
#define SEQ     2048
#define NBATCH  2
#define NHEADS  12
#define DK      64
#define TOKENS  (NBATCH * SEQ)             // 4096
#define HSIZE   ((size_t)TOKENS * D_MODEL) // 3,145,728 bf16 elems per activation plane
#define WSZ     ((size_t)D_MODEL * D_MODEL)

typedef __attribute__((ext_vector_type(16))) __bf16 v16bf;
typedef __attribute__((ext_vector_type(8)))  float  v8f;

union Frag16 {
    v16bf v;
    unsigned u[8];
    uint4 q[2];
};

// hardware f32->bf16 (RNE) pack: elem0 low half, elem1 high half
__device__ __forceinline__ unsigned pk2(float a, float b) {
    union { unsigned u; __bf16 h[2]; } r;
    r.h[0] = (__bf16)a;
    r.h[1] = (__bf16)b;
    return r.u;
}
__device__ __forceinline__ unsigned short bf1(float a) {
    union { unsigned short u; __bf16 h; } r;
    r.h = (__bf16)a;
    return r.u;
}
__device__ __forceinline__ v8f vzero() {
    v8f z = {0.f, 0.f, 0.f, 0.f, 0.f, 0.f, 0.f, 0.f};
    return z;
}
__device__ __forceinline__ v8f wmma_bf16(v16bf a, v16bf b, v8f c) {
    return __builtin_amdgcn_wmma_f32_16x16x32_bf16(false, a, false, b, (short)0, c,
                                                   false, false);
}

// A fragment (16x32 bf16): per-lane K pattern {koff+0..7, koff+16..23};
// caller pre-offsets p by koff = (lane>>4)*8.
__device__ __forceinline__ void loadA_bf16(Frag16& A, const unsigned short* p) {
    A.q[0] = *(const uint4*)(p);
    A.q[1] = *(const uint4*)(p + 16);
}
// B fragment (32x16 bf16): per-lane K pattern {koff+0..7, koff+16..23};
// caller pre-offsets p by koff = (lane>>4)*8.
__device__ __forceinline__ void loadB_bf16(Frag16& B, const unsigned short* p) {
    B.q[0] = *(const uint4*)(p);
    B.q[1] = *(const uint4*)(p + 16);
}

// ---------------------------------------------------------------------------
// Tensor Data Mover: 2-D tile (tile_w x tile_h bf16, row stride in elements)
// Global -> LDS. Descriptor per CDNA5 ISA D# layout (count=1, type=2,
// data_size=2B, tensor_dim == tile_dim, stride0 = row stride).
// Fallback: wave-cooperative copy if the builtin is unavailable.
// ---------------------------------------------------------------------------
typedef __attribute__((ext_vector_type(4))) unsigned int tdm_u32x4;
typedef __attribute__((ext_vector_type(8))) int          tdm_i32x8;
typedef __attribute__((ext_vector_type(4))) int          tdm_i32x4;

#if __has_builtin(__builtin_amdgcn_tensor_load_to_lds)
#define HAVE_TDM 1
#else
#define HAVE_TDM 0
#endif

__device__ __forceinline__ void tdm_load_2d(unsigned short* lds_dst,
                                            const unsigned short* gsrc,
                                            int tile_w, int tile_h, int row_stride,
                                            int lane) {
#if HAVE_TDM
    const unsigned long long ga = (unsigned long long)(size_t)gsrc;
    const unsigned la = (unsigned)(size_t)lds_dst;   // flat LDS addr low bits = offset
    tdm_u32x4 g0;
    g0[0] = 1u;                                                 // count=1 (user mode)
    g0[1] = la;                                                 // lds_addr (bytes)
    g0[2] = (unsigned)ga;                                       // global_addr[31:0]
    g0[3] = (unsigned)((ga >> 32) & 0x1FFFFFFu) | (2u << 30);   // addr[56:32] | type=2
    tdm_i32x8 g1;
    g1[0] = 1 << 16;                                            // data_size = 2 bytes
    g1[1] = (int)(((unsigned)tile_w & 0xFFFFu) << 16);          // tensor_dim0 lo
    g1[2] = (int)((((unsigned)tile_w >> 16) & 0xFFFFu) |
                  (((unsigned)tile_h & 0xFFFFu) << 16));        // tdim0 hi | tdim1 lo
    g1[3] = (int)((((unsigned)tile_h >> 16) & 0xFFFFu) |
                  (((unsigned)tile_w & 0xFFFFu) << 16));        // tdim1 hi | tile_dim0
    g1[4] = (int)((unsigned)tile_h & 0xFFFFu);                  // tile_dim1 (tile_dim2=0)
    g1[5] = row_stride;                                         // tensor_dim0_stride
    g1[6] = 0;
    g1[7] = 0;
    tdm_i32x4 gz = {0, 0, 0, 0};
#if defined(__clang_major__) && __clang_major__ >= 23
    tdm_i32x8 gz8 = {0, 0, 0, 0, 0, 0, 0, 0};
    __builtin_amdgcn_tensor_load_to_lds(g0, g1, gz, gz, gz8, 0);
#else
    __builtin_amdgcn_tensor_load_to_lds(g0, g1, gz, gz, 0);
#endif
#else
    const int w8 = tile_w >> 3;
    for (int idx = lane; idx < tile_h * w8; idx += 32) {
        const int r = idx / w8, c = (idx % w8) * 8;
        *(uint4*)(lds_dst + r * tile_w + c) = *(const uint4*)(gsrc + r * row_stride + c);
    }
#endif
}

__device__ __forceinline__ void tdm_wait_prev() {
#if HAVE_TDM && __has_builtin(__builtin_amdgcn_s_wait_tensorcnt)
    __builtin_amdgcn_s_wait_tensorcnt(2);   // allow the 2 just-issued to stay in flight
#endif
}
__device__ __forceinline__ void tdm_wait_all() {
#if HAVE_TDM && __has_builtin(__builtin_amdgcn_s_wait_tensorcnt)
    __builtin_amdgcn_s_wait_tensorcnt(0);
#endif
}

// ---------------------------------------------------------------------------
// One-shot f32 -> bf16 conversion (8 elems/thread).
// ---------------------------------------------------------------------------
__global__ __launch_bounds__(256) void k_cvt(const float* __restrict__ in,
                                             unsigned short* __restrict__ out, int n8) {
    const int i = blockIdx.x * 256 + threadIdx.x;
    if (i >= n8) return;
    const float4* p = (const float4*)in + 2 * (size_t)i;
    float4 a = p[0], b = p[1];
    uint4 st;
    st.x = pk2(a.x, a.y);
    st.y = pk2(a.z, a.w);
    st.z = pk2(b.x, b.y);
    st.w = pk2(b.z, b.w);
    ((uint4*)out)[i] = st;
}

// ---------------------------------------------------------------------------
// Q/K projection: Y = X @ W^T + b, bf16 in/out row-major [token, 768].
// A = W rows (M = out-feature), B = X rows (N = token). Pipelined.
// ---------------------------------------------------------------------------
__global__ __launch_bounds__(256) void k_proj_qk(const unsigned short* __restrict__ x,
                                                 const unsigned short* __restrict__ w,
                                                 const float* __restrict__ bias,
                                                 unsigned short* __restrict__ out) {
    const int lane = threadIdx.x & 31;
    const int wv   = threadIdx.x >> 5;
    const int lm   = lane & 15, lh = lane >> 4;
    const int f0   = blockIdx.y * 128 + wv * 16;
    const int t0   = blockIdx.x * 64;

    v8f acc[4] = {vzero(), vzero(), vzero(), vzero()};
    const unsigned short* wrow = w + (size_t)(f0 + lm) * D_MODEL + lh * 8;
    const unsigned short* x0   = x + (size_t)(t0 + lm) * D_MODEL + lh * 8;

    Frag16 Ab[2], Bb[2][4];
    loadA_bf16(Ab[0], wrow);
#pragma unroll
    for (int j = 0; j < 4; ++j) loadB_bf16(Bb[0][j], x0 + (size_t)(16 * j) * D_MODEL);

#pragma unroll 2
    for (int k = 0; k < D_MODEL - 32; k += 32) {
        const int cur = (k >> 5) & 1, nxt = cur ^ 1;
        loadA_bf16(Ab[nxt], wrow + k + 32);
#pragma unroll
        for (int j = 0; j < 4; ++j)
            loadB_bf16(Bb[nxt][j], x0 + (size_t)(16 * j) * D_MODEL + k + 32);
        __builtin_prefetch(wrow + k + 96, 0, 1);
        __builtin_prefetch(x0 + k + 96, 0, 1);
#pragma unroll
        for (int j = 0; j < 4; ++j)
            acc[j] = wmma_bf16(Ab[cur].v, Bb[cur][j].v, acc[j]);
    }
#pragma unroll
    for (int j = 0; j < 4; ++j)   // last k-step: parity (736>>5)&1 == 1
        acc[j] = wmma_bf16(Ab[1].v, Bb[1][j].v, acc[j]);

    float4 bb0 = *(const float4*)(bias + f0 + lh * 8);
    float4 bb1 = *(const float4*)(bias + f0 + lh * 8 + 4);
    float bv[8] = {bb0.x, bb0.y, bb0.z, bb0.w, bb1.x, bb1.y, bb1.z, bb1.w};
#pragma unroll
    for (int j = 0; j < 4; ++j) {
        uint4 st;
        st.x = pk2(acc[j][0] + bv[0], acc[j][1] + bv[1]);
        st.y = pk2(acc[j][2] + bv[2], acc[j][3] + bv[3]);
        st.z = pk2(acc[j][4] + bv[4], acc[j][5] + bv[5]);
        st.w = pk2(acc[j][6] + bv[6], acc[j][7] + bv[7]);
        const int tok = t0 + 16 * j + lm;
        *(uint4*)(out + (size_t)tok * D_MODEL + f0 + lh * 8) = st;
    }
}

// ---------------------------------------------------------------------------
// V projection, stored TRANSPOSED: Vt[b, h, d, s] bf16 (contiguous along s).
// A = X rows (M = token), B = W rows (N = out-feature). Pipelined.
// ---------------------------------------------------------------------------
__global__ __launch_bounds__(256) void k_proj_v(const unsigned short* __restrict__ x,
                                                const unsigned short* __restrict__ w,
                                                const float* __restrict__ bias,
                                                unsigned short* __restrict__ vt) {
    const int lane = threadIdx.x & 31;
    const int wv   = threadIdx.x >> 5;
    const int lm   = lane & 15, lh = lane >> 4;
    const int m0   = blockIdx.x * 128 + wv * 16;   // token base (never crosses batch)
    const int n0   = blockIdx.y * 64;              // feature base (head aligned)

    v8f acc[4] = {vzero(), vzero(), vzero(), vzero()};
    const unsigned short* xrow = x + (size_t)(m0 + lm) * D_MODEL + lh * 8;
    const unsigned short* w0   = w + (size_t)(n0 + lm) * D_MODEL + lh * 8;

    Frag16 Ab[2], Bb[2][4];
    loadA_bf16(Ab[0], xrow);
#pragma unroll
    for (int j = 0; j < 4; ++j) loadB_bf16(Bb[0][j], w0 + (size_t)(16 * j) * D_MODEL);

#pragma unroll 2
    for (int k = 0; k < D_MODEL - 32; k += 32) {
        const int cur = (k >> 5) & 1, nxt = cur ^ 1;
        loadA_bf16(Ab[nxt], xrow + k + 32);
#pragma unroll
        for (int j = 0; j < 4; ++j)
            loadB_bf16(Bb[nxt][j], w0 + (size_t)(16 * j) * D_MODEL + k + 32);
        __builtin_prefetch(xrow + k + 96, 0, 1);
        __builtin_prefetch(w0 + k + 96, 0, 1);
#pragma unroll
        for (int j = 0; j < 4; ++j)
            acc[j] = wmma_bf16(Ab[cur].v, Bb[cur][j].v, acc[j]);
    }
#pragma unroll
    for (int j = 0; j < 4; ++j)
        acc[j] = wmma_bf16(Ab[1].v, Bb[1][j].v, acc[j]);

#pragma unroll
    for (int j = 0; j < 4; ++j) {
        const int n     = n0 + 16 * j + lm;
        const float bvv = bias[n];
        const int h  = n >> 6, d = n & 63;
        const int bb = m0 >> 11;                   // / SEQ
        const int sr = (m0 & (SEQ - 1)) + lh * 8;
        uint4 st;
        st.x = pk2(acc[j][0] + bvv, acc[j][1] + bvv);
        st.y = pk2(acc[j][2] + bvv, acc[j][3] + bvv);
        st.z = pk2(acc[j][4] + bvv, acc[j][5] + bvv);
        st.w = pk2(acc[j][6] + bvv, acc[j][7] + bvv);
        *(uint4*)(vt + (size_t)((bb * NHEADS + h) * DK + d) * SEQ + sr) = st;
    }
}

// ---------------------------------------------------------------------------
// Flash attention. All 8 waves of a block share (batch, head) and consume the
// SAME K/V tiles, so wave 0 drives the Tensor Data Mover: K tile (32x64 bf16)
// and V tile (64x32 bf16) are DMA'd into double-buffered LDS while the block
// computes on the previous tile. Fragments come from LDS (ds_load_b128).
// S^T = K x Q^T orientation keeps the softmax'd P fragment directly usable as
// the A-operand of P x V (no cross-lane transpose).
// ---------------------------------------------------------------------------
__global__ __launch_bounds__(256) void k_attn(const unsigned short* __restrict__ Qh,
                                              const unsigned short* __restrict__ Kh,
                                              const unsigned short* __restrict__ Vt,
                                              unsigned short* __restrict__ Ctx) {
    __shared__ unsigned short sK[2][32 * DK];   // 32 keys x 64 d
    __shared__ unsigned short sV[2][DK * 32];   // 64 d x 32 keys

    const int lane = threadIdx.x & 31;
    const int wv   = threadIdx.x >> 5;
    const int gw   = blockIdx.x * 8 + wv;                     // 0..3071
    const int bb   = gw / (NHEADS * (SEQ / 16));
    const int rem  = gw % (NHEADS * (SEQ / 16));
    const int hh   = rem / (SEQ / 16);                        // same for all 8 waves
    const int s0   = (rem % (SEQ / 16)) * 16;
    const int lm   = lane & 15, lh = lane >> 4;
    const int sb   = lh * 8;  // shuffle source base for ctx-space rows

    Frag16 Bq0, Bq1;
    const unsigned short* qp =
        Qh + (size_t)(bb * SEQ + s0 + lm) * D_MODEL + hh * DK + lh * 8;
    loadB_bf16(Bq0, qp);
    loadB_bf16(Bq1, qp + 32);

    const unsigned short* kTile = Kh + (size_t)(bb * SEQ) * D_MODEL + hh * DK;
    const unsigned short* vTile = Vt + (size_t)((bb * NHEADS + hh) * DK) * SEQ;

    if (wv == 0) {  // stage tile tt=0 into buffer 0
        tdm_load_2d(&sK[0][0], kTile, DK, 32, D_MODEL, lane);
        tdm_load_2d(&sV[0][0], vTile, 32, DK, SEQ, lane);
    }

    float mrun = -1e30f, lrun = 0.f;
    v8f cx[4] = {vzero(), vzero(), vzero(), vzero()};

#pragma unroll 1
    for (int tt = 0; tt < SEQ; tt += 32) {
        const int cur = (tt >> 5) & 1, nxt = cur ^ 1;
        if (wv == 0) {
            if (tt + 32 < SEQ) {
                tdm_load_2d(&sK[nxt][0], kTile + (size_t)(tt + 32) * D_MODEL, DK, 32,
                            D_MODEL, lane);
                tdm_load_2d(&sV[nxt][0], vTile + (tt + 32), 32, DK, SEQ, lane);
                tdm_wait_prev();   // retire current tile's DMA, keep next in flight
            } else {
                tdm_wait_all();
            }
        }
        __syncthreads();           // current tile visible to all waves

        Frag16 Ak0, Ak1, Ak2, Ak3;
        loadA_bf16(Ak0, &sK[cur][lm * DK + lh * 8]);
        loadA_bf16(Ak1, &sK[cur][lm * DK + lh * 8 + 32]);
        loadA_bf16(Ak2, &sK[cur][(16 + lm) * DK + lh * 8]);
        loadA_bf16(Ak3, &sK[cur][(16 + lm) * DK + lh * 8 + 32]);

        v8f sc0 = vzero(), sc1 = vzero();
        sc0 = wmma_bf16(Ak0.v, Bq0.v, sc0);
        sc0 = wmma_bf16(Ak1.v, Bq1.v, sc0);
        sc1 = wmma_bf16(Ak2.v, Bq0.v, sc1);
        sc1 = wmma_bf16(Ak3.v, Bq1.v, sc1);

        float bm = -1e30f;
#pragma unroll
        for (int r = 0; r < 8; ++r) {
            sc0[r] *= 0.125f;
            sc1[r] *= 0.125f;
            bm = fmaxf(bm, fmaxf(sc0[r], sc1[r]));
        }
        bm = fmaxf(bm, __shfl_xor(bm, 16, 32));
        const float mnew  = fmaxf(mrun, bm);
        const float alpha = __expf(mrun - mnew);

        float rs = 0.f;
        Frag16 Ap;  // P as A-operand of the PV WMMA (in-register build)
#pragma unroll
        for (int r = 0; r < 4; ++r) {
            float e0 = __expf(sc0[2 * r]     - mnew);
            float e1 = __expf(sc0[2 * r + 1] - mnew);
            float e2 = __expf(sc1[2 * r]     - mnew);
            float e3 = __expf(sc1[2 * r + 1] - mnew);
            rs += (e0 + e1) + (e2 + e3);
            Ap.u[r]     = pk2(e0, e1);
            Ap.u[4 + r] = pk2(e2, e3);
        }
        rs += __shfl_xor(rs, 16, 32);
        lrun = lrun * alpha + rs;
        mrun = mnew;

        float ar[8];
#pragma unroll
        for (int r = 0; r < 8; ++r) ar[r] = __shfl(alpha, sb + r, 32);

#pragma unroll
        for (int j = 0; j < 4; ++j) {
            Frag16 Bv;
            loadB_bf16(Bv, &sV[cur][(16 * j + lm) * 32 + lh * 8]);
#pragma unroll
            for (int r = 0; r < 8; ++r) cx[j][r] *= ar[r];
            cx[j] = wmma_bf16(Ap.v, Bv.v, cx[j]);
        }
        __syncthreads();           // all reads done before buffer is overwritten
    }

    const float linv = 1.f / lrun;
    float lr[8];
#pragma unroll
    for (int r = 0; r < 8; ++r) lr[r] = __shfl(linv, sb + r, 32);

#pragma unroll
    for (int j = 0; j < 4; ++j) {
        const int feat = hh * DK + j * 16 + lm;
#pragma unroll
        for (int r = 0; r < 8; ++r) {
            const int tok = bb * SEQ + s0 + lh * 8 + r;
            Ctx[(size_t)tok * D_MODEL + feat] = bf1(cx[j][r] * lr[r]);
        }
    }
}

// ---------------------------------------------------------------------------
// Output projection: out = Ctx @ Wo^T + bo, f32 output. Pipelined.
// ---------------------------------------------------------------------------
__global__ __launch_bounds__(256) void k_oproj(const unsigned short* __restrict__ ctx,
                                               const unsigned short* __restrict__ w,
                                               const float* __restrict__ bias,
                                               float* __restrict__ out) {
    const int lane = threadIdx.x & 31;
    const int wv   = threadIdx.x >> 5;
    const int lm   = lane & 15, lh = lane >> 4;
    const int f0   = blockIdx.y * 128 + wv * 16;
    const int t0   = blockIdx.x * 64;

    v8f acc[4] = {vzero(), vzero(), vzero(), vzero()};
    const unsigned short* wrow = w + (size_t)(f0 + lm) * D_MODEL + lh * 8;
    const unsigned short* c0   = ctx + (size_t)(t0 + lm) * D_MODEL + lh * 8;

    Frag16 Ab[2], Bb[2][4];
    loadA_bf16(Ab[0], wrow);
#pragma unroll
    for (int j = 0; j < 4; ++j) loadB_bf16(Bb[0][j], c0 + (size_t)(16 * j) * D_MODEL);

#pragma unroll 2
    for (int k = 0; k < D_MODEL - 32; k += 32) {
        const int cur = (k >> 5) & 1, nxt = cur ^ 1;
        loadA_bf16(Ab[nxt], wrow + k + 32);
#pragma unroll
        for (int j = 0; j < 4; ++j)
            loadB_bf16(Bb[nxt][j], c0 + (size_t)(16 * j) * D_MODEL + k + 32);
        __builtin_prefetch(wrow + k + 96, 0, 1);
        __builtin_prefetch(c0 + k + 96, 0, 1);
#pragma unroll
        for (int j = 0; j < 4; ++j)
            acc[j] = wmma_bf16(Ab[cur].v, Bb[cur][j].v, acc[j]);
    }
#pragma unroll
    for (int j = 0; j < 4; ++j)
        acc[j] = wmma_bf16(Ab[1].v, Bb[1][j].v, acc[j]);

    float4 bb0 = *(const float4*)(bias + f0 + lh * 8);
    float4 bb1 = *(const float4*)(bias + f0 + lh * 8 + 4);
    float bv[8] = {bb0.x, bb0.y, bb0.z, bb0.w, bb1.x, bb1.y, bb1.z, bb1.w};
#pragma unroll
    for (int j = 0; j < 4; ++j) {
        const int tok = t0 + 16 * j + lm;
        float* op = out + (size_t)tok * D_MODEL + f0 + lh * 8;
        float4 s0v = {acc[j][0] + bv[0], acc[j][1] + bv[1],
                      acc[j][2] + bv[2], acc[j][3] + bv[3]};
        float4 s1v = {acc[j][4] + bv[4], acc[j][5] + bv[5],
                      acc[j][6] + bv[6], acc[j][7] + bv[7]};
        *(float4*)(op)     = s0v;
        *(float4*)(op + 4) = s1v;
    }
}

extern "C" void kernel_launch(void* const* d_in, const int* in_sizes, int n_in,
                              void* d_out, int out_size, void* d_ws, size_t ws_size,
                              hipStream_t stream) {
    const float* q   = (const float*)d_in[0];
    const float* k   = (const float*)d_in[1];
    const float* v   = (const float*)d_in[2];
    const float* w_q = (const float*)d_in[3];
    const float* b_q = (const float*)d_in[4];
    const float* w_k = (const float*)d_in[5];
    const float* b_k = (const float*)d_in[6];
    const float* w_v = (const float*)d_in[7];
    const float* b_v = (const float*)d_in[8];
    const float* w_o = (const float*)d_in[9];
    const float* b_o = (const float*)d_in[10];

    unsigned short* Qh  = (unsigned short*)d_ws;
    unsigned short* Kh  = Qh + HSIZE;
    unsigned short* Vt  = Kh + HSIZE;
    unsigned short* Ctx = Vt + HSIZE;
    unsigned short* Xq  = Ctx + HSIZE;
    unsigned short* Xk  = Xq + HSIZE;
    unsigned short* Xv  = Xk + HSIZE;
    unsigned short* Wq  = Xv + HSIZE;
    unsigned short* Wk  = Wq + WSZ;
    unsigned short* Wv  = Wk + WSZ;
    unsigned short* Wo  = Wv + WSZ;

    dim3 blk(256);
    const int nAct8 = (int)(HSIZE / 8);  // 393216 -> 1536 blocks
    const int nW8   = (int)(WSZ / 8);    // 73728  -> 288 blocks

    // one-shot f32 -> bf16 conversions
    k_cvt<<<dim3(nAct8 / 256), blk, 0, stream>>>(q, Xq, nAct8);
    k_cvt<<<dim3(nAct8 / 256), blk, 0, stream>>>(k, Xk, nAct8);
    k_cvt<<<dim3(nAct8 / 256), blk, 0, stream>>>(v, Xv, nAct8);
    k_cvt<<<dim3(nW8 / 256), blk, 0, stream>>>(w_q, Wq, nW8);
    k_cvt<<<dim3(nW8 / 256), blk, 0, stream>>>(w_k, Wk, nW8);
    k_cvt<<<dim3(nW8 / 256), blk, 0, stream>>>(w_v, Wv, nW8);
    k_cvt<<<dim3(nW8 / 256), blk, 0, stream>>>(w_o, Wo, nW8);

    // projections
    k_proj_qk<<<dim3(TOKENS / 64, D_MODEL / 128), blk, 0, stream>>>(Xq, Wq, b_q, Qh);
    k_proj_qk<<<dim3(TOKENS / 64, D_MODEL / 128), blk, 0, stream>>>(Xk, Wk, b_k, Kh);
    k_proj_v<<<dim3(TOKENS / 128, D_MODEL / 64), blk, 0, stream>>>(Xv, Wv, b_v, Vt);

    // attention: 3072 waves, 8 per block (one block shares one (b,h) K/V stream)
    k_attn<<<dim3((NBATCH * NHEADS * (SEQ / 16)) / 8), blk, 0, stream>>>(Qh, Kh, Vt, Ctx);

    // output projection -> f32 d_out
    k_oproj<<<dim3(TOKENS / 64, D_MODEL / 128), blk, 0, stream>>>(Ctx, Wo, b_o,
                                                                  (float*)d_out);
}
